// FrequencySparseAttention_24567212933792
// MI455X (gfx1250) — matrix operand order repaired and checked by reading, not compiled, not run on hardware
//
#include <hip/hip_runtime.h>

typedef __attribute__((ext_vector_type(16))) _Float16 v16h;
typedef __attribute__((ext_vector_type(8)))  _Float16 v8h;
typedef __attribute__((ext_vector_type(2)))  __fp16   v2fp;   // cvt_pkrtz return type
typedef __attribute__((ext_vector_type(8)))  float    v8f;

union Frag16 { v16h v; v8h h[2]; };

__device__ __forceinline__ v8f wmma_f16(const Frag16& a, const Frag16& b, v8f c) {
    // D = A(16x32 f16) x B(32x16 f16) + C(16x16 f32)
    return __builtin_amdgcn_wmma_f32_16x16x32_f16(
        false, a.v, false, b.v, (short)0, c, false, false);
}

// raw v_exp_f32 (2^x), no libm range fixup — fine for softmax (x <= 0)
__device__ __forceinline__ float fast_exp2(float x) {
    return __builtin_amdgcn_exp2f(x);
}

// pack 8 f32 -> 8 f16 with v_cvt_pk_rtz_f16_f32 (4 VALU ops)
__device__ __forceinline__ v8h cvt8(float4 x, float4 y) {
    union { v8h v; v2fp p[4]; } u;
    u.p[0] = __builtin_amdgcn_cvt_pkrtz(x.x, x.y);
    u.p[1] = __builtin_amdgcn_cvt_pkrtz(x.z, x.w);
    u.p[2] = __builtin_amdgcn_cvt_pkrtz(y.x, y.y);
    u.p[3] = __builtin_amdgcn_cvt_pkrtz(y.z, y.w);
    return u.v;
}

// ---- DPP 16-lane butterfly reductions (pure VALU, no LDS) -----------------
// xor1 = quad_perm[1,0,3,2] (0xB1), xor2 = quad_perm[2,3,0,1] (0x4E),
// then row_half_mirror (0x141) and row_mirror (0x140) complete the 16 lanes.
template<int CTRL>
__device__ __forceinline__ float dpp_mov(float x) {
    return __int_as_float(__builtin_amdgcn_update_dpp(
        0, __float_as_int(x), CTRL, 0xf, 0xf, true));
}
__device__ __forceinline__ float red_max16(float x) {
    x = fmaxf(x, dpp_mov<0xB1>(x));
    x = fmaxf(x, dpp_mov<0x4E>(x));
    x = fmaxf(x, dpp_mov<0x141>(x));
    x = fmaxf(x, dpp_mov<0x140>(x));
    return x;
}
__device__ __forceinline__ float red_sum16(float x) {
    x += dpp_mov<0xB1>(x);
    x += dpp_mov<0x4E>(x);
    x += dpp_mov<0x141>(x);
    x += dpp_mov<0x140>(x);
    return x;
}

// ---------------------------------------------------------------------------
// Gate: gate[k] = sigmoid((mean_j |ec[k,j]| - 0.1) * 10)
// ---------------------------------------------------------------------------
__global__ void gate_kernel(const float* __restrict__ ec, float* __restrict__ gate, int L) {
    int t = blockIdx.x * blockDim.x + threadIdx.x;
    if (t < L) {
        float s = 0.f;
        #pragma unroll
        for (int j = 0; j < 8; ++j) s += fabsf(ec[t * 8 + j]);
        s *= 0.125f;
        gate[t] = 1.f / (1.f + __expf(-(s - 0.1f) * 10.f));
    }
}

// ---------------------------------------------------------------------------
// WMMA GEMM:  C[M,N] = A[M,K] * W[K,N] + bias
// Block tile 128x64, BK=32, 256 threads = 8 waves (4x2), wave tile 32x32.
// A staged [m][k] f16; W staged transposed [n][k] f16 (B frags contiguous).
// ---------------------------------------------------------------------------
#define GM_BM 128
#define GM_BN 64
#define GM_BK 32

template<bool A_HALF, bool OUT_HALF>
__global__ __launch_bounds__(256)
void gemm_wmma(const void* __restrict__ Aram, const float* __restrict__ W,
               const float* __restrict__ bias, void* __restrict__ Cram,
               int M, int N, int K) {
    __shared__ _Float16 ldsA[GM_BM * GM_BK];   // [m][k], stride 32
    __shared__ _Float16 ldsB[GM_BN * GM_BK];   // [n][k], stride 32
    const int tid = threadIdx.x;
    const int m0 = blockIdx.x * GM_BM;
    const int n0 = blockIdx.y * GM_BN;
    const int w = tid >> 5, lane = tid & 31;
    const int wm = w >> 1, wn = w & 1;         // 4 (M) x 2 (N) waves
    const int hg = lane >> 4, lr = lane & 15;  // half-group, lane-in-group

    v8f acc[2][2] = {};

    const int rowA = (tid * 2) >> 2,  c8A = ((tid * 2) & 3) * 8;   // chunk 2*tid: rows 0-127, k 0/16
    const int kkW  = tid >> 3,        n8W = (tid & 7) * 8;

    for (int k0 = 0; k0 < K; k0 += GM_BK) {
        // prefetch next tiles into L2 while this block computes
        if (k0 + GM_BK < K) {
            if (A_HALF) {
                const _Float16* Ah = (const _Float16*)Aram;
                __builtin_prefetch(&Ah[(size_t)(m0 + rowA) * K + k0 + GM_BK + c8A], 0, 1);
            } else {
                const float* Af = (const float*)Aram;
                __builtin_prefetch(&Af[(size_t)(m0 + rowA) * K + k0 + GM_BK + c8A], 0, 1);
            }
            __builtin_prefetch(&W[(size_t)(k0 + GM_BK + kkW) * N + n0 + n8W], 0, 1);
        }
        // ---- stage A (128x32) ----
        #pragma unroll
        for (int i = 0; i < 2; ++i) {
            int idx = tid + i * 256;           // 512 chunks of 8 elems
            int row = idx >> 2, c8 = (idx & 3) * 8;
            if (A_HALF) {
                const _Float16* Ah = (const _Float16*)Aram;
                *(v8h*)&ldsA[row * GM_BK + c8] =
                    *(const v8h*)&Ah[(size_t)(m0 + row) * K + k0 + c8];
            } else {
                const float* Af = (const float*)Aram;
                const float4* src = (const float4*)&Af[(size_t)(m0 + row) * K + k0 + c8];
                *(v8h*)&ldsA[row * GM_BK + c8] = cvt8(src[0], src[1]);
            }
        }
        // ---- stage W transposed: ldsB[n][k] = W[k0+k][n0+n] ----
        {
            const float4* src = (const float4*)&W[(size_t)(k0 + kkW) * N + n0 + n8W];
            union { v8h v; _Float16 e[8]; } u;
            u.v = cvt8(src[0], src[1]);
            #pragma unroll
            for (int j = 0; j < 8; ++j)
                ldsB[(n8W + j) * GM_BK + kkW] = u.e[j];
        }
        __syncthreads();

        Frag16 a[2], b[2];
        #pragma unroll
        for (int ms = 0; ms < 2; ++ms) {
            int m = wm * 32 + ms * 16 + lr;
            a[ms].h[0] = *(const v8h*)&ldsA[m * GM_BK + 8 * hg];
            a[ms].h[1] = *(const v8h*)&ldsA[m * GM_BK + 16 + 8 * hg];
        }
        #pragma unroll
        for (int ns = 0; ns < 2; ++ns) {
            int n = wn * 32 + ns * 16 + lr;
            b[ns].h[0] = *(const v8h*)&ldsB[n * GM_BK + 8 * hg];
            b[ns].h[1] = *(const v8h*)&ldsB[n * GM_BK + 16 + 8 * hg];
        }
        #pragma unroll
        for (int ms = 0; ms < 2; ++ms)
            #pragma unroll
            for (int ns = 0; ns < 2; ++ns)
                acc[ms][ns] = wmma_f16(a[ms], b[ns], acc[ms][ns]);
        __syncthreads();
    }

    // ---- epilogue (+bias) ----
    #pragma unroll
    for (int ms = 0; ms < 2; ++ms) {
        #pragma unroll
        for (int ns = 0; ns < 2; ++ns) {
            int n = n0 + wn * 32 + ns * 16 + lr;
            float bv = bias[n];
            #pragma unroll
            for (int p = 0; p < 8; ++p) {
                int m = m0 + wm * 32 + ms * 16 + p + 8 * hg;
                float val = acc[ms][ns][p] + bv;
                if (OUT_HALF) ((_Float16*)Cram)[(size_t)m * N + n] = (_Float16)val;
                else          ((float*)Cram)[(size_t)m * N + n] = val;
            }
        }
    }
}

// ---------------------------------------------------------------------------
// Gated flash attention, one head per blockIdx.y, 128 queries per block.
// 8 waves; each wave owns 16 queries. 64 keys per iteration (4 S-tiles,
// 2-step P*V); row reductions are pure-VALU DPP butterflies.
// Softmax runs in the exp2 domain: Q is pre-scaled by 0.125*log2(e) at
// staging, so S needs no per-element scaling and exp() is a bare v_exp_f32.
// Online softmax: l accumulates UNGATED exp; gate folded into P before P*V.
// ---------------------------------------------------------------------------
#define AT_QB 128
#define AT_KB 64

__global__ __launch_bounds__(256)
void attn_kernel(const _Float16* __restrict__ Qh, const _Float16* __restrict__ Kh,
                 const _Float16* __restrict__ Vh, const float* __restrict__ gate,
                 _Float16* __restrict__ Oh, int L) {
    __shared__ _Float16 ldsQ[AT_QB * 64];        // [q][d]    16KB
    __shared__ _Float16 ldsK[AT_KB * 64];        // [key][d]   8KB
    __shared__ _Float16 ldsV[64 * AT_KB];        // [d][key]   8KB (transposed)
    __shared__ _Float16 ldsP[8][16 * AT_KB];     // per-wave P scratch, 16KB

    const int tid = threadIdx.x;
    const int h = blockIdx.y;
    const int qb0 = blockIdx.x * AT_QB;
    const int w = tid >> 5, lane = tid & 31;
    const int hg = lane >> 4, lr = lane & 15;
    const size_t hoff = (size_t)h * 64;

    // stage Q tile 128x64, pre-scaled by 1/sqrt(64) * log2(e)
    const _Float16 qscale = (_Float16)(0.125f * 1.44269504089f);
    #pragma unroll
    for (int i = 0; i < 4; ++i) {
        int idx = tid + i * 256;
        int row = idx >> 3, d8 = (idx & 7) * 8;
        v8h qv = *(const v8h*)&Qh[(size_t)(qb0 + row) * 1024 + hoff + d8];
        *(v8h*)&ldsQ[row * 64 + d8] = qv * qscale;
    }
    __syncthreads();

    Frag16 aQ[2];
    {
        int m = w * 16 + lr;
        #pragma unroll
        for (int ds = 0; ds < 2; ++ds) {
            aQ[ds].h[0] = *(const v8h*)&ldsQ[m * 64 + ds * 32 + 8 * hg];
            aQ[ds].h[1] = *(const v8h*)&ldsQ[m * 64 + ds * 32 + 16 + 8 * hg];
        }
    }

    v8f accO[4] = {};
    float mrun[8], lrun[8];
    #pragma unroll
    for (int p = 0; p < 8; ++p) { mrun[p] = -1e30f; lrun[p] = 0.f; }

    const int keyS = tid >> 3, d8S = (tid & 7) * 8;   // staging coords

    for (int kb = 0; kb < L; kb += AT_KB) {
        // stage K [key][d] and V transposed [d][key]; prefetch next tiles
        if (kb + AT_KB < L) {
            __builtin_prefetch(&Kh[(size_t)(kb + AT_KB + keyS) * 1024 + hoff + d8S], 0, 1);
            __builtin_prefetch(&Vh[(size_t)(kb + AT_KB + keyS) * 1024 + hoff + d8S], 0, 1);
        }
        #pragma unroll
        for (int i = 0; i < 2; ++i) {
            int idx = tid + i * 256;                  // 512 chunks of 8
            int key = idx >> 3, d8 = (idx & 7) * 8;
            *(v8h*)&ldsK[key * 64 + d8] =
                *(const v8h*)&Kh[(size_t)(kb + key) * 1024 + hoff + d8];
            v8h vv = *(const v8h*)&Vh[(size_t)(kb + key) * 1024 + hoff + d8];
            #pragma unroll
            for (int j = 0; j < 8; ++j)
                ldsV[(d8 + j) * AT_KB + key] = vv[j];
        }
        __syncthreads();

        // S = Q K^T : 4 key-subtiles x 2 d-steps (already in exp2 domain)
        v8f s[4];
        #pragma unroll
        for (int kn = 0; kn < 4; ++kn) {
            v8f c = {};
            #pragma unroll
            for (int ds = 0; ds < 2; ++ds) {
                Frag16 bK;
                int n = kn * 16 + lr;
                bK.h[0] = *(const v8h*)&ldsK[n * 64 + ds * 32 + 8 * hg];
                bK.h[1] = *(const v8h*)&ldsK[n * 64 + ds * 32 + 16 + 8 * hg];
                c = wmma_f16(aQ[ds], bK, c);
            }
            s[kn] = c;
        }

        float g[4];
        #pragma unroll
        for (int kn = 0; kn < 4; ++kn) g[kn] = gate[kb + kn * 16 + lr];

        float alpha[8];
        #pragma unroll
        for (int p = 0; p < 8; ++p) {
            float mb = fmaxf(fmaxf(s[0][p], s[1][p]), fmaxf(s[2][p], s[3][p]));
            mb = red_max16(mb);
            float mn = fmaxf(mrun[p], mb);
            float e[4], r = 0.f;
            #pragma unroll
            for (int kn = 0; kn < 4; ++kn) { e[kn] = fast_exp2(s[kn][p] - mn); r += e[kn]; }
            r = red_sum16(r);
            float a = fast_exp2(mrun[p] - mn);
            lrun[p] = lrun[p] * a + r;     // ungated denominator
            mrun[p] = mn;
            alpha[p] = a;
            int mrow = p + 8 * hg;         // C-layout row -> P A-layout row
            #pragma unroll
            for (int kn = 0; kn < 4; ++kn)
                ldsP[w][mrow * AT_KB + kn * 16 + lr] = (_Float16)(e[kn] * g[kn]);
        }

        #pragma unroll
        for (int ns = 0; ns < 4; ++ns)
            #pragma unroll
            for (int p = 0; p < 8; ++p)
                accO[ns][p] *= alpha[p];

        // P (16x64, gated) x V (64x64): 2 k-steps x 4 dh-subtiles
        Frag16 aP[2];
        #pragma unroll
        for (int ks = 0; ks < 2; ++ks) {
            aP[ks].h[0] = *(const v8h*)&ldsP[w][lr * AT_KB + ks * 32 + 8 * hg];
            aP[ks].h[1] = *(const v8h*)&ldsP[w][lr * AT_KB + ks * 32 + 16 + 8 * hg];
        }
        #pragma unroll
        for (int ns = 0; ns < 4; ++ns) {
            #pragma unroll
            for (int ks = 0; ks < 2; ++ks) {
                Frag16 bV;
                int n = ns * 16 + lr;
                bV.h[0] = *(const v8h*)&ldsV[n * AT_KB + ks * 32 + 8 * hg];
                bV.h[1] = *(const v8h*)&ldsV[n * AT_KB + ks * 32 + 16 + 8 * hg];
                accO[ns] = wmma_f16(aP[ks], bV, accO[ns]);
            }
        }
        __syncthreads();   // before next iteration overwrites ldsK/ldsV
    }

    // normalize by ungated softmax sum, store f16 [L, D] with head offset
    float rl[8];
    #pragma unroll
    for (int p = 0; p < 8; ++p) rl[p] = 1.f / lrun[p];
    #pragma unroll
    for (int ns = 0; ns < 4; ++ns) {
        int dcol = ns * 16 + lr;
        #pragma unroll
        for (int p = 0; p < 8; ++p) {
            int q = qb0 + w * 16 + p + 8 * hg;
            Oh[(size_t)q * 1024 + hoff + dcol] = (_Float16)(accO[ns][p] * rl[p]);
        }
    }
}

// ---------------------------------------------------------------------------
// Launch: gate -> Q/K/V projections (f32 in, f16 out) -> attention -> out-proj
// Workspace: Qh/Kh/Vh/Ah (8MB each, f16) + gate (16KB)  ~= 32MB of d_ws.
// ---------------------------------------------------------------------------
extern "C" void kernel_launch(void* const* d_in, const int* in_sizes, int n_in,
                              void* d_out, int out_size, void* d_ws, size_t ws_size,
                              hipStream_t stream) {
    const int L = 4096, D = 1024;
    const float* q_x = (const float*)d_in[0];
    const float* k_x = (const float*)d_in[1];
    const float* v_x = (const float*)d_in[2];
    const float* ec  = (const float*)d_in[3];
    const float* Wq  = (const float*)d_in[4];
    const float* bq  = (const float*)d_in[5];
    const float* Wk  = (const float*)d_in[6];
    const float* bk  = (const float*)d_in[7];
    const float* Wv  = (const float*)d_in[8];
    const float* bv  = (const float*)d_in[9];
    const float* Wo  = (const float*)d_in[10];
    const float* bo  = (const float*)d_in[11];
    float* out = (float*)d_out;

    char* ws = (char*)d_ws;
    const size_t halfMat = (size_t)L * D * sizeof(_Float16);   // 8 MB
    _Float16* Qh  = (_Float16*)(ws);
    _Float16* Kh  = (_Float16*)(ws + halfMat);
    _Float16* Vh  = (_Float16*)(ws + 2 * halfMat);
    _Float16* Ah  = (_Float16*)(ws + 3 * halfMat);
    float*    gateBuf = (float*)(ws + 4 * halfMat);

    gate_kernel<<<(L + 255) / 256, 256, 0, stream>>>(ec, gateBuf, L);

    dim3 ggrid(L / GM_BM, D / GM_BN);   // 32 x 16
    gemm_wmma<false, true><<<ggrid, 256, 0, stream>>>(q_x, Wq, bq, Qh, L, D, D);
    gemm_wmma<false, true><<<ggrid, 256, 0, stream>>>(k_x, Wk, bk, Kh, L, D, D);
    gemm_wmma<false, true><<<ggrid, 256, 0, stream>>>(v_x, Wv, bv, Vh, L, D, D);

    dim3 agrid(L / AT_QB, 16);          // 32 query-blocks x 16 heads
    attn_kernel<<<agrid, 256, 0, stream>>>(Qh, Kh, Vh, gateBuf, Ah, L);

    gemm_wmma<true, false><<<ggrid, 256, 0, stream>>>(Ah, Wo, bo, out, L, D, D);
}
